// CrossGraphAttention_25039659336451
// MI455X (gfx1250) — compile-verified, matrix operand order checked by reading, not run on hardware
//
#include <hip/hip_runtime.h>
#include <hip/hip_bf16.h>

typedef __attribute__((ext_vector_type(16))) _Float16 v16h;
typedef __attribute__((ext_vector_type(8)))  _Float16 v8h;
typedef __attribute__((ext_vector_type(8)))  float    v8f;

#define N_NODES 50000
#define C_IN    19
#define C_OUT   128
#define M_TILES 3125        // 50000 / 16 exactly
#define GEMM_WAVES 5        // 3125 = 625 blocks * 5 waves exactly -> no guards, EXEC all-1s

// ---------------------------------------------------------------------------
// Kernel 1 (fused both branches): xt_{h,k} = x @ W_{h,k}.T + b_{h,k}
// via v_wmma_f32_16x16x32_f16.  x rows and W are staged into zero-padded
// (K: 19->32) f16 LDS so fragment builds are two ds_load_b128 per operand,
// with no divergent guards anywhere near the WMMAs.
// ---------------------------------------------------------------------------
__global__ __launch_bounds__(32 * GEMM_WAVES)
void gemm_wmma_kernel(const float* __restrict__ x,
                      const float* __restrict__ Wh, const float* __restrict__ bh,
                      const float* __restrict__ Wk, const float* __restrict__ bk,
                      float* __restrict__ xt_h, float* __restrict__ xt_k) {
    __shared__ __align__(16) _Float16 Wl[2][C_OUT * 32];        // 2 x 8KB, K padded
    __shared__ __align__(16) _Float16 Al[GEMM_WAVES][16 * 32];  // 1KB per wave, K padded

    const int tid  = threadIdx.x;
    const int wave = tid >> 5;
    const int lid  = tid & 31;
    const int m0   = (blockIdx.x * GEMM_WAVES + wave) * 16;     // exact: always < 50000

    // Zero-fill padded LDS (covers K = 19..31)
    uint32_t* wz = (uint32_t*)&Wl[0][0];
    for (int i = tid; i < 2 * C_OUT * 16; i += 32 * GEMM_WAVES) wz[i] = 0;
    uint32_t* az = (uint32_t*)&Al[wave][0];
    for (int i = lid; i < 16 * 16; i += 32) az[i] = 0;
    __syncthreads();

    // Stage both W matrices (128x19 each) -> f16, contiguous global reads
    for (int i = tid; i < C_OUT * C_IN; i += 32 * GEMM_WAVES) {
        const int col = i / C_IN, K = i - col * C_IN;
        Wl[0][col * 32 + K] = (_Float16)Wh[i];
        Wl[1][col * 32 + K] = (_Float16)Wk[i];
    }
    // Stage this wave's 16 rows of x (16x19 floats, contiguous)
    for (int i = lid; i < 16 * C_IN; i += 32) {
        const int r = i / C_IN, K = i - r * C_IN;
        Al[wave][r * 32 + K] = (_Float16)x[m0 * C_IN + i];
    }
    __syncthreads();

    const int half = lid >> 4;     // A/B f16 layout: lanes 0-15 K{0..7,16..23}, 16-31 K{8..15,24..31}
    const int l    = lid & 15;

    // A fragment: two 16B LDS chunks
    const v8h* ap = (const v8h*)&Al[wave][l * 32];
    const v8h alo = ap[half];
    const v8h ahi = ap[2 + half];
    v16h a;
#pragma unroll
    for (int e = 0; e < 8; ++e) { a[e] = alo[e]; a[e + 8] = ahi[e]; }

#pragma unroll
    for (int br = 0; br < 2; ++br) {
        const float* bias = (br == 0) ? bh : bk;
        float* xt = (br == 0) ? xt_h : xt_k;
#pragma unroll
        for (int t = 0; t < 8; ++t) {
            const int col = t * 16 + l;
            const v8h* bp = (const v8h*)&Wl[br][col * 32];
            const v8h blo = bp[half];
            const v8h bhi = bp[2 + half];
            v16h b;
#pragma unroll
            for (int e = 0; e < 8; ++e) { b[e] = blo[e]; b[e + 8] = bhi[e]; }

            const float bv = bias[col];
            v8f c;
#pragma unroll
            for (int v = 0; v < 8; ++v) c[v] = bv;   // bias folded into accumulator

            c = __builtin_amdgcn_wmma_f32_16x16x32_f16(
                    false, a, false, b, (short)0, c, false, false);
#pragma unroll
            for (int v = 0; v < 8; ++v)              // D: row = v + 8*half, col = l
                xt[(m0 + v + 8 * half) * C_OUT + col] = c[v];
        }
    }
}

// ---------------------------------------------------------------------------
// Kernel 2: per-node attention scalars (separable edge logit)
//   p[n] = xt[n,:] . aW[0:128] + ab      q[n] = xt[n,:] . aW[128:256]
// ---------------------------------------------------------------------------
__global__ void node_scalars_kernel(const float* __restrict__ xt,
                                    const float* __restrict__ aW,
                                    const float* __restrict__ ab,
                                    float* __restrict__ p,
                                    float* __restrict__ q) {
    const int wave = threadIdx.x >> 5;
    const int lid  = threadIdx.x & 31;
    const int n    = blockIdx.x * (blockDim.x >> 5) + wave;
    if (n >= N_NODES) return;
    float accp = 0.0f, accq = 0.0f;
#pragma unroll
    for (int i = 0; i < 4; ++i) {
        const int c = lid + 32 * i;
        const float v = xt[n * C_OUT + c];
        accp += v * aW[c];
        accq += v * aW[C_OUT + c];
    }
#pragma unroll
    for (int m = 16; m >= 1; m >>= 1) {
        accp += __shfl_xor(accp, m, 32);
        accq += __shfl_xor(accq, m, 32);
    }
    if (lid == 0) { p[n] = accp + ab[0]; q[n] = accq; }
}

// ---------------------------------------------------------------------------
// Kernel 3: edge scatter (runtime dominator; xt/out are L2-resident).
//   s = sigmoid(p[dst] + q[src]);  out[dst,:] += s * xt[src,:]
// ---------------------------------------------------------------------------
__global__ void edge_scatter_kernel(const float* __restrict__ xt,
                                    const float* __restrict__ p,
                                    const float* __restrict__ q,
                                    const int* __restrict__ src,
                                    const int* __restrict__ dst,
                                    float* __restrict__ out,
                                    int n_edges) {
    const int wave = threadIdx.x >> 5;
    const int lid  = threadIdx.x & 31;
    const int e    = blockIdx.x * (blockDim.x >> 5) + wave;
    if (e >= n_edges) return;
    const int s = src[e];
    const int d = dst[e];
    const float att = 1.0f / (1.0f + expf(-(p[d] + q[s])));
    const float4 v = reinterpret_cast<const float4*>(xt + (size_t)s * C_OUT)[lid];
    float* o = out + (size_t)d * C_OUT + 4 * lid;
    atomicAdd(o + 0, att * v.x);
    atomicAdd(o + 1, att * v.y);
    atomicAdd(o + 2, att * v.z);
    atomicAdd(o + 3, att * v.w);
}

// ---------------------------------------------------------------------------
// Kernel 4: gate combine (in place over hout == d_out).
// ---------------------------------------------------------------------------
__global__ void gate_kernel(float* __restrict__ hout,
                            const float* __restrict__ kout,
                            const float* __restrict__ gW,   // [2,256] row-major
                            const float* __restrict__ gb) { // [2]
    const int wave = threadIdx.x >> 5;
    const int lid  = threadIdx.x & 31;
    const int n    = blockIdx.x * (blockDim.x >> 5) + wave;
    if (n >= N_NODES) return;
    const float4 h = reinterpret_cast<const float4*>(hout + (size_t)n * C_OUT)[lid];
    const float4 k = reinterpret_cast<const float4*>(kout + (size_t)n * C_OUT)[lid];
    const int c = 4 * lid;
    float s0 = h.x * gW[c]       + h.y * gW[c + 1]       + h.z * gW[c + 2]       + h.w * gW[c + 3]
             + k.x * gW[128 + c] + k.y * gW[128 + c + 1] + k.z * gW[128 + c + 2] + k.w * gW[128 + c + 3];
    float s1 = h.x * gW[256 + c] + h.y * gW[256 + c + 1] + h.z * gW[256 + c + 2] + h.w * gW[256 + c + 3]
             + k.x * gW[384 + c] + k.y * gW[384 + c + 1] + k.z * gW[384 + c + 2] + k.w * gW[384 + c + 3];
#pragma unroll
    for (int m = 16; m >= 1; m >>= 1) {
        s0 += __shfl_xor(s0, m, 32);
        s1 += __shfl_xor(s1, m, 32);
    }
    const float g0 = 1.0f / (1.0f + expf(-(s0 + gb[0])));
    const float g1 = 1.0f / (1.0f + expf(-(s1 + gb[1])));
    float4 r;
    r.x = g0 * h.x + g1 * k.x;
    r.y = g0 * h.y + g1 * k.y;
    r.z = g0 * h.z + g1 * k.z;
    r.w = g0 * h.w + g1 * k.w;
    reinterpret_cast<float4*>(hout + (size_t)n * C_OUT)[lid] = r;
}

extern "C" void kernel_launch(void* const* d_in, const int* in_sizes, int n_in,
                              void* d_out, int out_size, void* d_ws, size_t ws_size,
                              hipStream_t stream) {
    const float* x       = (const float*)d_in[0];
    const int*   h_edge  = (const int*)  d_in[1];  // [2,E]: row0=src, row1=dst
    const int*   k_edge  = (const int*)  d_in[2];
    const float* h_W     = (const float*)d_in[3];
    const float* h_b     = (const float*)d_in[4];
    const float* h_aW    = (const float*)d_in[5];
    const float* h_ab    = (const float*)d_in[6];
    const float* k_W     = (const float*)d_in[7];
    const float* k_b     = (const float*)d_in[8];
    const float* k_aW    = (const float*)d_in[9];
    const float* k_ab    = (const float*)d_in[10];
    const float* g_W     = (const float*)d_in[11];
    const float* g_b     = (const float*)d_in[12];

    const int E_h = in_sizes[1] / 2;
    const int E_k = in_sizes[2] / 2;

    // Workspace layout (floats), ~77.6 MB total
    float* ws    = (float*)d_ws;
    float* xt_h  = ws;
    float* xt_k  = ws + (size_t)N_NODES * C_OUT;
    float* out_k = ws + (size_t)2 * N_NODES * C_OUT;
    float* p_h   = ws + (size_t)3 * N_NODES * C_OUT;
    float* q_h   = p_h + N_NODES;
    float* p_k   = q_h + N_NODES;
    float* q_k   = p_k + N_NODES;

    float* out_h = (float*)d_out;   // hyper accumulator, then final output

    hipMemsetAsync(out_h, 0, (size_t)N_NODES * C_OUT * sizeof(float), stream);
    hipMemsetAsync(out_k, 0, (size_t)N_NODES * C_OUT * sizeof(float), stream);

    // 1) Fused feature-lift GEMMs via WMMA (exact tiling: 625 blocks x 5 waves)
    gemm_wmma_kernel<<<M_TILES / GEMM_WAVES, 32 * GEMM_WAVES, 0, stream>>>(
        x, h_W, h_b, k_W, k_b, xt_h, xt_k);

    // 2) Per-node attention scalars
    dim3 node_grid((N_NODES + 7) / 8);
    node_scalars_kernel<<<node_grid, 256, 0, stream>>>(xt_h, h_aW, h_ab, p_h, q_h);
    node_scalars_kernel<<<node_grid, 256, 0, stream>>>(xt_k, k_aW, k_ab, p_k, q_k);

    // 3) Edge gather/scale/scatter (wave per edge)
    edge_scatter_kernel<<<(E_h + 7) / 8, 256, 0, stream>>>(
        xt_h, p_h, q_h, h_edge, h_edge + E_h, out_h, E_h);
    edge_scatter_kernel<<<(E_k + 7) / 8, 256, 0, stream>>>(
        xt_k, p_k, q_k, k_edge, k_edge + E_k, out_k, E_k);

    // 4) Gate and combine in place into d_out
    gate_kernel<<<node_grid, 256, 0, stream>>>(out_h, out_k, g_W, g_b);
}
